// SparseNodeModel_57595511439648
// MI455X (gfx1250) — matrix operand order repaired and checked by reading, not compiled, not run on hardware
//
#include <hip/hip_runtime.h>

typedef __attribute__((ext_vector_type(16))) _Float16 v16h;
typedef __attribute__((ext_vector_type(8)))  _Float16 v8h;
typedef __attribute__((ext_vector_type(8)))  float    v8f;
typedef __attribute__((ext_vector_type(4)))  float    v4f;

#define BATCH   2048
#define IN_DIM  4096
#define OUT_DIM 1024
#define N_EDGES 16384

#define BM 64
#define BN 128
#define BK 32
#define APITCH 40   // halfs per A-tile row (32 + 8 pad, keeps 16B align, spreads banks)
#define BPITCH 40   // halfs per B^T-tile row

// ---------- kernel 1: zero the dense weight matrix (workspace) ----------
__global__ void wann_zero_w(float* __restrict__ W) {
  W[(size_t)blockIdx.x * 256 + threadIdx.x] = 0.0f;
}

// ---------- kernel 2: scatter edges into dense W[src][dst] ----------
__global__ void wann_scatter(const float* __restrict__ wts,
                             const int* __restrict__ esrc,
                             const int* __restrict__ edst,
                             float* __restrict__ W) {
  int e = blockIdx.x * 256 + threadIdx.x;
  if (e < N_EDGES) {
    atomicAdd(&W[(size_t)esrc[e] * OUT_DIM + edst[e]], wts[e]);
  }
}

// ---------- kernel 3: W f32 [K][N] -> f16 transposed WhT [N][K] ----------
__global__ void wann_transpose_w(const float* __restrict__ W,
                                 _Float16* __restrict__ WhT) {
  int t  = blockIdx.x * 256 + threadIdx.x;   // 0 .. OUT_DIM*(IN_DIM/8)-1
  int n  = t & (OUT_DIM - 1);
  int k8 = (t >> 10) * 8;
  v8h h;
#pragma unroll
  for (int j = 0; j < 8; ++j) h[j] = (_Float16)W[(size_t)(k8 + j) * OUT_DIM + n];
  *(v8h*)&WhT[(size_t)n * IN_DIM + k8] = h;
}

// sigmoid(tanh(s)) via hardware exp2/rcp (v_exp_f32, v_rcp_f32), branch-free.
__device__ __forceinline__ float sig_tanh_fast(float s) {
  const float L2E = 1.4426950408889634f;
  float e1 = __builtin_amdgcn_exp2f(2.0f * L2E * s);
  float t  = 1.0f - 2.0f * __builtin_amdgcn_rcpf(1.0f + e1);
  float e2 = __builtin_amdgcn_exp2f(-L2E * t);
  return __builtin_amdgcn_rcpf(1.0f + e2);
}

// issue one async 16B global->LDS copy (per-lane addresses), ASYNCcnt-tracked
__device__ __forceinline__ void async_b128_to_lds(const _Float16* gsrc, _Float16* ldst) {
  unsigned           lds = (unsigned)(size_t)ldst;        // low 32 bits = LDS offset
  unsigned long long ga  = (unsigned long long)gsrc;
  asm volatile("global_load_async_to_lds_b128 %0, %1, off"
               :: "v"(lds), "v"(ga) : "memory");
}
__device__ __forceinline__ void wait_async0() {
  asm volatile("s_wait_asynccnt 0" ::: "memory");
}

// ---------- kernel 4: fused GEMM (WMMA f16->f32) + tanh + sigmoid ----------
__global__ __launch_bounds__(256)
void wann_gemm_act(const float* __restrict__ X,
                   const _Float16* __restrict__ WhT,
                   float* __restrict__ Out) {
  // double-buffered tiles: A row-major [m][k], B transposed [n][k]
  __shared__ __align__(16) _Float16 Al[2][BM * APITCH];
  __shared__ __align__(16) _Float16 Bl[2][BN * BPITCH];

  const int t    = threadIdx.x;
  const int lane = t & 31;
  const int wid  = t >> 5;        // 0..7 waves
  const int waveM = wid & 3;      // 4 waves down M (16 rows each)
  const int waveN = wid >> 2;     // 2 waves across N (64 cols each)
  const int m0 = blockIdx.x * BM;
  const int n0 = blockIdx.y * BN;

  // A staging: thread covers row ar, 8 contiguous k's at ac
  const int ar  = t >> 2;          // 0..63
  const int ac  = (t & 3) * 8;     // 0,8,16,24
  // B staging: thread covers column bn, 16 contiguous k's at bkh (two 16B asyncs)
  const int bn  = t & 127;         // 0..127
  const int bkh = (t >> 7) * 16;   // 0 or 16
  const _Float16* wrow = WhT + (size_t)(n0 + bn) * IN_DIM + bkh;

  // WMMA fragment addressing (ISA 7.12.2, wave32)
  const int hsel = lane >> 4;
  const int lidx = lane & 15;
  const int kA   = hsel * 8;
  const int kB   = hsel * 8;
  const int aRow = waveM * 16 + lidx;

  v8f acc[4] = {};
  v4f x0, x1;

  // ---- prologue: stage tile 0 ----
  {
    const float* xp = X + (size_t)(m0 + ar) * IN_DIM + ac;
    x0 = *(const v4f*)xp;
    x1 = *(const v4f*)(xp + 4);
    async_b128_to_lds(wrow,     &Bl[0][bn * BPITCH + bkh]);
    async_b128_to_lds(wrow + 8, &Bl[0][bn * BPITCH + bkh + 8]);
    v8h a8;
#pragma unroll
    for (int j = 0; j < 4; ++j) { a8[j] = (_Float16)x0[j]; a8[4 + j] = (_Float16)x1[j]; }
    *(v8h*)&Al[0][ar * APITCH + ac] = a8;
    wait_async0();
  }
  __syncthreads();

  int buf = 0;
  for (int kb = 0; kb < IN_DIM; kb += BK) {
    const bool hasNext = (kb + BK) < IN_DIM;
    const int  nb = buf ^ 1;

    // ---- issue next tile's loads early (overlap with WMMA below) ----
    if (hasNext) {
      const int kn = kb + BK;
      // B tile: two async 16B copies straight into the other LDS buffer
      async_b128_to_lds(wrow + kn,     &Bl[nb][bn * BPITCH + bkh]);
      async_b128_to_lds(wrow + kn + 8, &Bl[nb][bn * BPITCH + bkh + 8]);
      // A tile: f32 loads into registers (converted at store time)
      const float* xp = X + (size_t)(m0 + ar) * IN_DIM + kn + ac;
      x0 = *(const v4f*)xp;
      x1 = *(const v4f*)(xp + 4);
    }

    // ---- hoist ALL fragment LDS loads, then the 4-WMMA chain ----
    // (10 in-order ds_load_b128 outstanding lets the scheduler use staggered
    //  partial s_wait_dscnt values instead of waiting to 0 before each WMMA)
    v8h alo = *(const v8h*)&Al[buf][aRow * APITCH + kA];
    v8h ahi = *(const v8h*)&Al[buf][aRow * APITCH + 16 + kA];
    v8h bfr[4][2];
#pragma unroll
    for (int f = 0; f < 4; ++f) {
      const int nrow = waveN * 64 + f * 16 + lidx;
      bfr[f][0] = *(const v8h*)&Bl[buf][nrow * BPITCH + kB];
      bfr[f][1] = *(const v8h*)&Bl[buf][nrow * BPITCH + 16 + kB];
    }

    v16h afrag;
#pragma unroll
    for (int j = 0; j < 8; ++j) { afrag[j] = alo[j]; afrag[8 + j] = ahi[j]; }

#pragma unroll
    for (int f = 0; f < 4; ++f) {
      v16h bfrag;
#pragma unroll
      for (int j = 0; j < 8; ++j) { bfrag[j] = bfr[f][0][j]; bfrag[8 + j] = bfr[f][1][j]; }
      acc[f] = __builtin_amdgcn_wmma_f32_16x16x32_f16(
          false, afrag, false, bfrag, (short)0, acc[f], false, false);
    }

    // ---- store next A tile, wait async B, single barrier ----
    if (hasNext) {
      v8h a8;
#pragma unroll
      for (int j = 0; j < 4; ++j) { a8[j] = (_Float16)x0[j]; a8[4 + j] = (_Float16)x1[j]; }
      *(v8h*)&Al[nb][ar * APITCH + ac] = a8;
      wait_async0();
      __syncthreads();
      buf = nb;
    }
  }

  // ---- epilogue: sigmoid(tanh(s)) with hardware transcendentals ----
  const int rowBase = m0 + waveM * 16 + hsel * 8;
#pragma unroll
  for (int f = 0; f < 4; ++f) {
    const int col = n0 + waveN * 64 + f * 16 + lidx;
#pragma unroll
    for (int r = 0; r < 8; ++r) {
      Out[(size_t)(rowBase + r) * OUT_DIM + col] = sig_tanh_fast(acc[f][r]);
    }
  }
}

extern "C" void kernel_launch(void* const* d_in, const int* in_sizes, int n_in,
                              void* d_out, int out_size, void* d_ws, size_t ws_size,
                              hipStream_t stream) {
  const float* x    = (const float*)d_in[0];   // [2048, 4096] f32
  const float* wts  = (const float*)d_in[1];   // [16384] f32
  const int*   esrc = (const int*)d_in[2];     // [16384] i32
  const int*   edst = (const int*)d_in[3];     // [16384] i32
  float* out = (float*)d_out;                  // [2048, 1024] f32

  float*    W   = (float*)d_ws;                           // [4096,1024] f32, 16 MB
  _Float16* WhT = (_Float16*)((char*)d_ws + (size_t)IN_DIM * OUT_DIM * 4); // [1024,4096] f16, 8 MB

  wann_zero_w<<<(IN_DIM * OUT_DIM) / 256, 256, 0, stream>>>(W);
  wann_scatter<<<(N_EDGES + 255) / 256, 256, 0, stream>>>(wts, esrc, edst, W);
  wann_transpose_w<<<(OUT_DIM * (IN_DIM / 8)) / 256, 256, 0, stream>>>(W, WhT);

  dim3 grid(BATCH / BM, OUT_DIM / BN);   // 32 x 8 = 256 workgroups
  wann_gemm_act<<<grid, 256, 0, stream>>>(x, WhT, out);

  (void)in_sizes; (void)n_in; (void)out_size; (void)ws_size;
}
